// CounterPropagationNetwork_17008070492287
// MI455X (gfx1250) — compile-verified, matrix operand order checked by reading, not run on hardware
//
#include <hip/hip_runtime.h>

#define B_SZ  32768
#define DIN   256
#define H_SZ  4096
#define DOUT  256
#define LDSW  264                  // 256 halves + 8 pad halves -> rotates banks per row
#define TILEB (16 * LDSW * 2)      // bytes per LDS tile buffer (8448)
#define NT    (H_SZ / 16)          // 256 N-tiles

typedef __attribute__((ext_vector_type(16))) _Float16 v16h;
typedef __attribute__((ext_vector_type(8)))  _Float16 v8h;
typedef __attribute__((ext_vector_type(8)))  float    v8f;

// ---------------------------------------------------------------------------
// CDNA5 async-copy helpers (inline asm: portable across both toolchains)
// ---------------------------------------------------------------------------
__device__ __forceinline__ void async_cp16(unsigned lds_byte_off, const void* gaddr) {
  // LDS[lds_byte_off .. +15] = MEM[gaddr .. +15], tracked by ASYNCcnt
  asm volatile("global_load_async_to_lds_b128 %0, %1, off"
               :: "v"(lds_byte_off), "v"(gaddr) : "memory");
}
__device__ __forceinline__ void wait_async0() {
  asm volatile("s_wait_asynccnt 0x0" ::: "memory");
}

// ---------------------------------------------------------------------------
// Kernel 0: one-time W f32 -> f16 conversion (2 MB workspace)
// ---------------------------------------------------------------------------
__global__ void cpn_wconv_kernel(const float* __restrict__ W, _Float16* __restrict__ Wh) {
  int i = (blockIdx.x * 256 + threadIdx.x) * 8;
  float4 a = *(const float4*)(W + i);
  float4 b = *(const float4*)(W + i + 4);
  v8h h;
  h[0] = (_Float16)a.x; h[1] = (_Float16)a.y; h[2] = (_Float16)a.z; h[3] = (_Float16)a.w;
  h[4] = (_Float16)b.x; h[5] = (_Float16)b.y; h[6] = (_Float16)b.z; h[7] = (_Float16)b.w;
  *(v8h*)(Wh + i) = h;
}

// ---------------------------------------------------------------------------
// Kernel 1: w2[h] = sum_k kohonen[h,k]^2
// ---------------------------------------------------------------------------
__global__ void cpn_w2_kernel(const float* __restrict__ W, float* __restrict__ w2) {
  int h = blockIdx.x * blockDim.x + threadIdx.x;
  if (h >= H_SZ) return;
  const float4* r = (const float4*)(W + (size_t)h * DIN);
  float s = 0.f;
#pragma unroll 8
  for (int i = 0; i < DIN / 4; ++i) {
    float4 v = r[i];
    s += v.x * v.x + v.y * v.y + v.z * v.z + v.w * v.w;
  }
  w2[h] = s;
}

// ---------------------------------------------------------------------------
// Kernel 2: GT[h,j] = G[j,h]  (tiled LDS transpose)
// ---------------------------------------------------------------------------
__global__ void cpn_transpose_kernel(const float* __restrict__ G, float* __restrict__ GT) {
  __shared__ float tile[32][33];
  int x = blockIdx.x * 32 + threadIdx.x;
  int y = blockIdx.y * 32 + threadIdx.y;
#pragma unroll
  for (int j = 0; j < 32; j += 8)
    tile[threadIdx.y + j][threadIdx.x] = G[(size_t)(y + j) * H_SZ + x];
  __syncthreads();
  int x2 = blockIdx.y * 32 + threadIdx.x;
  int y2 = blockIdx.x * 32 + threadIdx.y;
#pragma unroll
  for (int j = 0; j < 32; j += 8)
    GT[(size_t)(y2 + j) * DOUT + x2] = tile[threadIdx.x][threadIdx.y + j];
}

// ---------------------------------------------------------------------------
// Kernel 3: fused  S = x @ W^T (f16 WMMA, f32 accum)  +  argmin_h(w2[h]-2*S)
// Block = 128 threads = 4 waves; each wave owns TWO 16-row strips (128 rows/WG,
// grid = 256 blocks). Shared W tile double-buffered in LDS, filled by
// global_load_async_to_lds_b128 overlapped with WMMA on the other buffer.
// Each B fragment read from LDS feeds 2 independent WMMA chains.
// ---------------------------------------------------------------------------
__global__ __launch_bounds__(128) void cpn_argmin_kernel(
    const float* __restrict__ x, const _Float16* __restrict__ Wh,
    const float* __restrict__ w2, int* __restrict__ winners,
    float* __restrict__ winF) {
  __shared__ __align__(16) _Float16 wtile[2][16 * LDSW];

  const int tid   = threadIdx.x;
  const int lane  = tid & 31;      // wave32
  const int wave  = tid >> 5;      // 0..3
  const int lidx  = lane & 15;
  const int khalf = lane >> 4;     // elems0-7: K {0..7 | 8..15}; elems8-15: +16
  const int m0    = blockIdx.x * 128 + wave * 32;

  // ---- A fragments: 2 strips x 16 rows x 256 K, f32 -> f16 in registers ----
  v16h afrag[2][8];
#pragma unroll
  for (int s = 0; s < 2; ++s) {
    const float* xrow = x + (size_t)(m0 + s * 16 + lidx) * DIN;
#pragma unroll
    for (int kk = 0; kk < 8; ++kk) {
      const int kb = kk * 32 + khalf * 8;
      float4 a0 = *(const float4*)(xrow + kb);
      float4 a1 = *(const float4*)(xrow + kb + 4);
      float4 a2 = *(const float4*)(xrow + kb + 16);
      float4 a3 = *(const float4*)(xrow + kb + 20);
      v16h a;
      a[0]  = (_Float16)a0.x; a[1]  = (_Float16)a0.y; a[2]  = (_Float16)a0.z; a[3]  = (_Float16)a0.w;
      a[4]  = (_Float16)a1.x; a[5]  = (_Float16)a1.y; a[6]  = (_Float16)a1.z; a[7]  = (_Float16)a1.w;
      a[8]  = (_Float16)a2.x; a[9]  = (_Float16)a2.y; a[10] = (_Float16)a2.z; a[11] = (_Float16)a2.w;
      a[12] = (_Float16)a3.x; a[13] = (_Float16)a3.y; a[14] = (_Float16)a3.z; a[15] = (_Float16)a3.w;
      afrag[s][kk] = a;
    }
  }

  float minv[2][8];
  int   mini[2][8];
#pragma unroll
  for (int s = 0; s < 2; ++s)
#pragma unroll
    for (int j = 0; j < 8; ++j) { minv[s][j] = 3.4e38f; mini[s][j] = 0; }

  // LDS byte offset of the tile buffers (flat addr low 32 bits == LDS offset)
  const unsigned ldsbase = (unsigned)(unsigned long long)(void*)&wtile[0][0];

  // async stage of one 16x256 f16 W tile: 512 x 16B segments, 4 per thread
  auto issue_tile = [&](int nt, int bufsel) {
    const _Float16* gbase = Wh + (size_t)nt * 16 * DIN;
#pragma unroll
    for (int i = 0; i < 4; ++i) {
      int idx = i * 128 + tid;
      int row = idx >> 5, seg = idx & 31;
      async_cp16(ldsbase + (unsigned)(bufsel * TILEB + row * (LDSW * 2) + seg * 16),
                 gbase + row * DIN + seg * 8);
    }
  };

  issue_tile(0, 0);

  for (int nt = 0; nt < NT; ++nt) {
    const int cur = nt & 1;
    const int n0  = nt * 16;

    wait_async0();     // this wave's copies into buf[cur] have landed
    __syncthreads();   // everyone's copies visible; buf[1-cur] fully consumed
    if (nt + 1 < NT) issue_tile(nt + 1, 1 - cur);  // overlap with compute below

    // ---- 2 x (16x16x256): 8 shared B frags, 16 chained v_wmma ----
    v8f c0 = {}, c1 = {};
    const _Float16* wr = &wtile[cur][lidx * LDSW];
#pragma unroll
    for (int kk = 0; kk < 8; ++kk) {
      const int kb = kk * 32 + khalf * 8;
      v8h b0 = *(const v8h*)(wr + kb);
      v8h b1 = *(const v8h*)(wr + kb + 16);
      v16h bfrag;
#pragma unroll
      for (int e = 0; e < 8; ++e) { bfrag[e] = b0[e]; bfrag[e + 8] = b1[e]; }
      c0 = __builtin_amdgcn_wmma_f32_16x16x32_f16(
          false, afrag[0][kk], false, bfrag, (short)0, c0, false, false);
      c1 = __builtin_amdgcn_wmma_f32_16x16x32_f16(
          false, afrag[1][kk], false, bfrag, (short)0, c1, false, false);
    }

    // ---- running argmin of  w2[n] - 2*S  (x2 term is row-constant) ----
    const float w2v  = w2[n0 + lidx];
    const int   nIdx = n0 + lidx;
#pragma unroll
    for (int j = 0; j < 8; ++j) {
      float s0 = fmaf(-2.0f, c0[j], w2v);
      if (s0 < minv[0][j]) { minv[0][j] = s0; mini[0][j] = nIdx; }
      float s1 = fmaf(-2.0f, c1[j], w2v);
      if (s1 < minv[1][j]) { minv[1][j] = s1; mini[1][j] = nIdx; }
    }
  }

  // ---- butterfly min-reduce over the 16-lane N groups ----
#pragma unroll
  for (int s = 0; s < 2; ++s) {
#pragma unroll
    for (int j = 0; j < 8; ++j) {
      float v  = minv[s][j];
      int   id = mini[s][j];
#pragma unroll
      for (int m = 1; m < 16; m <<= 1) {
        float ov  = __shfl_xor(v, m, 32);
        int   oid = __shfl_xor(id, m, 32);
        if (ov < v || (ov == v && oid < id)) { v = ov; id = oid; }
      }
      if (lidx == 0) {  // lanes 0 & 16: rows (s*16)+j and (s*16)+8+j
        int row = m0 + s * 16 + khalf * 8 + j;
        winners[row] = id;
        winF[row]    = (float)id;
      }
    }
  }
}

// ---------------------------------------------------------------------------
// Kernel 4: out[b,:] = GT[winner[b],:]  (coalesced 1KB row copies)
// ---------------------------------------------------------------------------
__global__ void cpn_gather_kernel(const float* __restrict__ GT,
                                  const int* __restrict__ winners,
                                  float* __restrict__ out) {
  int row = blockIdx.x * 4 + (threadIdx.x >> 6);
  int j   = threadIdx.x & 63;
  int w   = winners[row];
  ((float4*)(out + (size_t)row * DOUT))[j] = ((const float4*)(GT + (size_t)w * DOUT))[j];
}

// ---------------------------------------------------------------------------
extern "C" void kernel_launch(void* const* d_in, const int* in_sizes, int n_in,
                              void* d_out, int out_size, void* d_ws, size_t ws_size,
                              hipStream_t stream) {
  const float* x  = (const float*)d_in[0];   // [B, DIN]
  const float* kw = (const float*)d_in[1];   // [H, DIN]
  const float* gw = (const float*)d_in[2];   // [DOUT, H]

  float* outF = (float*)d_out;                 // [B, DOUT]
  float* winF = outF + (size_t)B_SZ * DOUT;    // [B] winner idx as float

  char*      ws      = (char*)d_ws;
  float*     w2      = (float*)ws;                                         // 16 KB
  float*     GT      = (float*)(ws + 16 * 1024);                           // 4 MB
  int*       winners = (int*)(ws + 16 * 1024 + (size_t)H_SZ * DOUT * 4);   // 128 KB
  _Float16*  Wh      = (_Float16*)(ws + 16 * 1024 + (size_t)H_SZ * DOUT * 4
                                      + (size_t)B_SZ * 4);                 // 2 MB

  cpn_wconv_kernel<<<(H_SZ * DIN) / (256 * 8), 256, 0, stream>>>(kw, Wh);
  cpn_w2_kernel<<<H_SZ / 256, 256, 0, stream>>>(kw, w2);

  dim3 tb(32, 8), tg(H_SZ / 32, DOUT / 32);
  cpn_transpose_kernel<<<tg, tb, 0, stream>>>(gw, GT);

  cpn_argmin_kernel<<<B_SZ / 128, 128, 0, stream>>>(x, Wh, w2, winners, winF);

  cpn_gather_kernel<<<B_SZ / 4, 256, 0, stream>>>(GT, winners, outF);
}
